// ReverseLSTMLayer_74071005987573
// MI455X (gfx1250) — matrix operand order repaired and checked by reading, not compiled, not run on hardware
//
#include <hip/hip_runtime.h>

#define SEQ 2048
#define BB  32
#define II  256
#define HH  256
#define NWG 16        // workgroups, one hidden-slice each
#define JW  16        // hidden units per WG
#define NC  (4*JW)    // gate columns per WG = 64
#define THREADS 256   // 8 waves of 32

typedef __attribute__((ext_vector_type(8)))  __bf16 v8bf;
typedef __attribute__((ext_vector_type(16))) __bf16 v16bf;
typedef __attribute__((ext_vector_type(8)))  float  v8f;
typedef __attribute__((ext_vector_type(4)))  unsigned int uint32x4;
typedef __attribute__((ext_vector_type(8)))  int int32x8;
typedef __attribute__((ext_vector_type(4)))  int int32x4;

// ---- workspace layout (bytes) ----
#define OFF_WIH  256
#define OFF_WHH  (OFF_WIH + 524288)
#define OFF_BIAS (OFF_WHH + 524288)
#define OFF_HBUF (OFF_BIAS + 4096)
#define OFF_XBF  (OFF_HBUF + 16384)

__device__ __forceinline__ unsigned short f2bf(float f) {
  unsigned u = __float_as_uint(f);
  u += 0x7FFFu + ((u >> 16) & 1u);          // round-to-nearest-even
  return (unsigned short)(u >> 16);
}

typedef __attribute__((address_space(3))) void lds_void;
__device__ __forceinline__ unsigned lds_addr_of(void* p) {
  return (unsigned)(uintptr_t)(lds_void*)p;   // generic -> LDS byte offset
}

// TDM: 1-D DMA of `n8` 8-byte elements from global -> LDS (ISA ch.8 D# layout).
__device__ __forceinline__ void tdm_load_1d(unsigned lds_byte,
                                            unsigned long long gaddr,
                                            unsigned n8) {
  uint32x4 g0;
  g0[0] = 1u;                                   // count=1, user mode
  g0[1] = lds_byte;                             // lds_addr
  g0[2] = (unsigned)(gaddr & 0xFFFFFFFFull);    // global_addr[31:0]
  g0[3] = (unsigned)(gaddr >> 32) | (2u << 30); // global_addr[56:32] | type=2
  int32x8 g1;
  g1[0] = (int)(3u << 16);                      // data_size=3 (8B), no multicast
  g1[1] = (int)((n8 & 0xFFFFu) << 16);          // tensor_dim0[15:0]
  g1[2] = (int)((n8 >> 16) | (1u << 16));       // tensor_dim0[31:16] | tensor_dim1=1
  g1[3] = (int)((n8 & 0xFFFFu) << 16);          // tile_dim0 (n8 <= 65535)
  g1[4] = 0;                                    // tile_dim1=0, tile_dim2=0
  g1[5] = (int)n8;                              // tensor_dim0_stride[31:0]
  g1[6] = 0;
  g1[7] = 0;
  int32x4 z4 = {0, 0, 0, 0};
  int32x8 z8 = {0, 0, 0, 0, 0, 0, 0, 0};
  __builtin_amdgcn_tensor_load_to_lds(g0, g1, z4, z4, z8, 0);
}

// A fragment (16-bit 16x32, wave32): lane L holds row m0+(L&15);
// lanes 0-15: K = k0+0..7 and k0+16..23 ; lanes 16-31: +8 on both.
__device__ __forceinline__ v16bf load_A(const unsigned short* s, int ldk,
                                        int m0, int k0, int lane) {
  int m  = m0 + (lane & 15);
  int kb = k0 + ((lane >> 4) << 3);
  const __bf16* p = (const __bf16*)(s + m * ldk + kb);
  v8bf lo = *(const v8bf*)p;
  v8bf hi = *(const v8bf*)(p + 16);
  v16bf r;
#pragma unroll
  for (int i = 0; i < 8; ++i) { r[i] = lo[i]; r[i + 8] = hi[i]; }
  return r;
}

// B fragment (32x16): lane L holds column n0+(L&15); W row-major == B^T.
__device__ __forceinline__ v16bf load_B(const unsigned short* s, int ldk,
                                        int n0, int k0, int lane) {
  int n  = n0 + (lane & 15);
  int kb = k0 + ((lane >> 4) << 3);
  const __bf16* p = (const __bf16*)(s + n * ldk + kb);
  v8bf lo = *(const v8bf*)p;
  v8bf hi = *(const v8bf*)(p + 16);
  v16bf r;
#pragma unroll
  for (int i = 0; i < 8; ++i) { r[i] = lo[i]; r[i + 8] = hi[i]; }
  return r;
}

__global__ void lstm_setup(const float* __restrict__ x,
                           const float* __restrict__ W_ih, const float* __restrict__ W_hh,
                           const float* __restrict__ b_ih, const float* __restrict__ b_hh,
                           const float* __restrict__ h0,
                           unsigned short* __restrict__ wih_b,
                           unsigned short* __restrict__ whh_b,
                           float* __restrict__ bias,
                           unsigned short* __restrict__ h_buf,
                           unsigned short* __restrict__ xbf,
                           unsigned* __restrict__ bar) {
  const int NW = 4 * HH * II;  // 262144 per weight matrix
  long long idx = (long long)blockIdx.x * blockDim.x + threadIdx.x;
  if (idx < NW) {
    wih_b[idx] = f2bf(W_ih[idx]);
  } else if (idx < 2 * NW) {
    whh_b[idx - NW] = f2bf(W_hh[idx - NW]);
  } else if (idx < 2 * NW + 4 * HH) {
    int i = (int)(idx - 2 * NW);
    bias[i] = b_ih[i] + b_hh[i];
  } else if (idx < 2 * NW + 4 * HH + BB * HH) {
    int i = (int)(idx - 2 * NW - 4 * HH);
    h_buf[i] = f2bf(h0[i]);
  } else if (idx < 2LL * NW + 4 * HH + BB * HH + (long long)SEQ * BB * II) {
    long long i = idx - 2 * NW - 4 * HH - BB * HH;
    xbf[i] = f2bf(x[i]);
  }
  if (idx == 0) { bar[0] = 0u; bar[1] = 0u; }
}

__global__ void __launch_bounds__(THREADS, 1)
lstm_persistent(const unsigned short* __restrict__ xbf,
                const float* __restrict__ c0,
                const unsigned short* __restrict__ wih_g,
                const unsigned short* __restrict__ whh_g,
                const float* __restrict__ bias_g,
                unsigned short* __restrict__ h_buf,
                unsigned* __restrict__ bar,
                float* __restrict__ out) {
  __shared__ __align__(16) unsigned short Wih_s[NC * II];     // 32 KB
  __shared__ __align__(16) unsigned short Whh_s[NC * HH];     // 32 KB
  __shared__ __align__(16) unsigned short x_s[2][BB * II];    // 32 KB (double buffer)
  __shared__ __align__(16) unsigned short h_s[BB * HH];       // 16 KB
  __shared__ float gates_s[BB][NC];                           // 8 KB
  __shared__ float c_s[BB][JW];                               // 2 KB
  __shared__ float bias_s[NC];

  const int tid  = threadIdx.x;
  const int wg   = blockIdx.x;
  const int lane = tid & 31;
  const int wave = tid >> 5;

  // ---- one-time: weight slices into LDS (bf16, row-major) ----
  for (int c = tid; c < NC * (II / 8); c += THREADS) {
    int r  = c >> 5;
    int k8 = (c & 31) * 8;
    int grow = (r >> 4) * HH + wg * JW + (r & 15);
    *(uint4*)&Wih_s[r * II + k8] = *(const uint4*)&wih_g[grow * II + k8];
    *(uint4*)&Whh_s[r * HH + k8] = *(const uint4*)&whh_g[grow * HH + k8];
  }
  if (tid < NC)
    bias_s[tid] = bias_g[(tid >> 4) * HH + wg * JW + (tid & 15)];
#pragma unroll
  for (int pp = 0; pp < 2; ++pp) {
    int p = tid + pp * THREADS;
    int b = p >> 4, jj = p & 15;
    c_s[b][jj] = c0[b * HH + wg * JW + jj];
  }

  // prefetch x_{T-1} into buffer 0 via TDM
  if (wave == 0) {
    tdm_load_1d(lds_addr_of(x_s[0]),
                (unsigned long long)(uintptr_t)(xbf + (size_t)(SEQ - 1) * BB * II),
                (BB * II * 2) / 8);
  }
  __syncthreads();

  const int m0 = (wave & 1) * 16;
  const int n0 = (wave >> 1) * 16;

  for (int t = SEQ - 1; t >= 0; --t) {
    const int buf = ((SEQ - 1) - t) & 1;

    // ---- TDM: fetch h (just exchanged) + prefetch x_{t-1} into other buffer ----
    if (wave == 0) {
      tdm_load_1d(lds_addr_of(h_s),
                  (unsigned long long)(uintptr_t)h_buf, (BB * HH * 2) / 8);
      if (t > 0) {
        tdm_load_1d(lds_addr_of(x_s[buf ^ 1]),
                    (unsigned long long)(uintptr_t)(xbf + (size_t)(t - 1) * BB * II),
                    (BB * II * 2) / 8);
        __builtin_amdgcn_s_wait_tensorcnt(1);  // x_t + h done; x_{t-1} may fly
      } else {
        __builtin_amdgcn_s_wait_tensorcnt(0);
      }
    }
    __syncthreads();

    // ---- gates tile [16x16] = x_t @ Wih^T + h @ Whh^T (K = 256 + 256) ----
    v8f acc = (v8f)(0.0f);
#pragma unroll
    for (int kb = 0; kb < II / 32; ++kb) {
      v16bf a  = load_A(x_s[buf], II, m0, kb * 32, lane);
      v16bf bm = load_B(Wih_s, II, n0, kb * 32, lane);
      acc = __builtin_amdgcn_wmma_f32_16x16x32_bf16(false, a, false, bm,
                                                    (short)0, acc, false, false);
    }
#pragma unroll
    for (int kb = 0; kb < HH / 32; ++kb) {
      v16bf a  = load_A(h_s, HH, m0, kb * 32, lane);
      v16bf bm = load_B(Whh_s, HH, n0, kb * 32, lane);
      acc = __builtin_amdgcn_wmma_f32_16x16x32_bf16(false, a, false, bm,
                                                    (short)0, acc, false, false);
    }
    {
      int n  = lane & 15;
      int mb = (lane >> 4) * 8;
#pragma unroll
      for (int r = 0; r < 8; ++r)
        gates_s[m0 + mb + r][n0 + n] = acc[r];
    }
    __syncthreads();

    // ---- elementwise gates + state update for this WG's 16 hidden units ----
#pragma unroll
    for (int pp = 0; pp < 2; ++pp) {
      int p = tid + pp * THREADS;
      int b = p >> 4, jj = p & 15;
      float ig = gates_s[b][jj]          + bias_s[jj];
      float fg = gates_s[b][JW + jj]     + bias_s[JW + jj];
      float gg = gates_s[b][2 * JW + jj] + bias_s[2 * JW + jj];
      float og = gates_s[b][3 * JW + jj] + bias_s[3 * JW + jj];
      ig = 1.0f / (1.0f + __expf(-ig));
      fg = 1.0f / (1.0f + __expf(-fg));
      gg = tanhf(gg);
      og = 1.0f / (1.0f + __expf(-og));
      float cv = fg * c_s[b][jj] + ig * gg;
      c_s[b][jj] = cv;
      float hv = og * tanhf(cv);
      int j = wg * JW + jj;
      out[(size_t)t * BB * HH + b * HH + j] = hv;   // ys[t] (f32)
      h_buf[b * HH + j] = f2bf(hv);                 // bf16 exchange for next step
      if (t == 0) {
        out[(size_t)SEQ * BB * HH + b * HH + j]           = hv;  // hT
        out[(size_t)SEQ * BB * HH + BB * HH + b * HH + j] = cv;  // cT
      }
    }

    // ---- device-scope sense-reversing barrier across the 16 WGs ----
    __threadfence();
    __syncthreads();
    if (tid == 0) {
      unsigned g0 = __hip_atomic_load(&bar[1], __ATOMIC_RELAXED, __HIP_MEMORY_SCOPE_AGENT);
      unsigned old = __hip_atomic_fetch_add(&bar[0], 1u, __ATOMIC_ACQ_REL, __HIP_MEMORY_SCOPE_AGENT);
      if (old == NWG - 1) {
        __hip_atomic_store(&bar[0], 0u, __ATOMIC_RELAXED, __HIP_MEMORY_SCOPE_AGENT);
        __hip_atomic_fetch_add(&bar[1], 1u, __ATOMIC_RELEASE, __HIP_MEMORY_SCOPE_AGENT);
      } else {
        while (__hip_atomic_load(&bar[1], __ATOMIC_ACQUIRE, __HIP_MEMORY_SCOPE_AGENT) == g0) {
          __builtin_amdgcn_s_sleep(1);
        }
      }
    }
    __syncthreads();
  }
}

extern "C" void kernel_launch(void* const* d_in, const int* in_sizes, int n_in,
                              void* d_out, int out_size, void* d_ws, size_t ws_size,
                              hipStream_t stream) {
  const float* x    = (const float*)d_in[0];
  const float* h0   = (const float*)d_in[1];
  const float* c0   = (const float*)d_in[2];
  const float* W_ih = (const float*)d_in[3];
  const float* W_hh = (const float*)d_in[4];
  const float* b_ih = (const float*)d_in[5];
  const float* b_hh = (const float*)d_in[6];
  float* out = (float*)d_out;

  char* ws = (char*)d_ws;
  unsigned*       bar   = (unsigned*)ws;
  unsigned short* wih_b = (unsigned short*)(ws + OFF_WIH);
  unsigned short* whh_b = (unsigned short*)(ws + OFF_WHH);
  float*          bias  = (float*)(ws + OFF_BIAS);
  unsigned short* h_buf = (unsigned short*)(ws + OFF_HBUF);
  unsigned short* xbf   = (unsigned short*)(ws + OFF_XBF);

  const long long total = 2LL * (4 * HH * II) + 4 * HH + BB * HH + (long long)SEQ * BB * II;
  const int blocks = (int)((total + 255) / 256);
  lstm_setup<<<blocks, 256, 0, stream>>>(x, W_ih, W_hh, b_ih, b_hh, h0,
                                         wih_b, whh_b, bias, h_buf, xbf, bar);
  lstm_persistent<<<NWG, THREADS, 0, stream>>>(xbf, c0, wih_b, whh_b, bias,
                                               h_buf, bar, out);
}
